// VQAttention_33414845563604
// MI455X (gfx1250) — compile-verified, matrix operand order checked by reading, not run on hardware
//
#include <hip/hip_runtime.h>
#include <hip/hip_bf16.h>
#include <stdint.h>

// Problem sizes (fixed by the reference)
#define Bn 4
#define Hn 8
#define Ln 4096
#define Dn 128
#define Sn 1024

constexpr float kGamma = 0.99f;
constexpr float kEps   = 0.01f;
constexpr float kOmG   = 1.0f - kGamma;   // (1 - C_GAMMA)
constexpr float kNScl  = 1.0f;            // N_DEVICE * N_BLOCK

typedef __attribute__((ext_vector_type(16))) __bf16 v16bf;
typedef __attribute__((ext_vector_type(8)))  float  v8f;

__device__ __forceinline__ unsigned short f2bf_rne(float x) {
  unsigned int u = __float_as_uint(x);
  u = u + 0x7FFFu + ((u >> 16) & 1u);
  return (unsigned short)(u >> 16);
}

// CDNA5 async copy global->LDS (VGLOBAL op 98, tracked by ASYNCcnt).
// dst: wave-relative LDS byte address; src: 64-bit global address.
__device__ __forceinline__ void async_g2l_b128(unsigned lds_addr, const void* gptr) {
  asm volatile("global_load_async_to_lds_b128 %0, %1, off"
               :: "v"(lds_addr), "v"((unsigned long long)(uintptr_t)gptr)
               : "memory");
}

__device__ __forceinline__ void wait_async0() {
#if __has_builtin(__builtin_amdgcn_s_wait_asynccnt)
  __builtin_amdgcn_s_wait_asynccnt(0);
#else
  asm volatile("s_wait_asynccnt 0x0" ::: "memory");
#endif
}

// ---------------------------------------------------------------------------
// Kernel A0: zero the loss accumulators and the n_tok scratch scalar
// ---------------------------------------------------------------------------
__global__ void k_init(float* commit_slot, float* ntok) {
  commit_slot[0] = 0.0f;   // l_commit
  commit_slot[1] = 0.0f;   // l_codebook
  ntok[0]        = 0.0f;
}

// ---------------------------------------------------------------------------
// Kernel A1: n_tok = sum(loss_mask)   (B*L = 16384 = 64 blocks * 256)
// ---------------------------------------------------------------------------
__global__ void k_ntok(const int* __restrict__ mask, float* __restrict__ ntok) {
  int i = blockIdx.x * 256 + threadIdx.x;
  float v = (i < Bn * Ln) ? (float)mask[i] : 0.0f;
  #pragma unroll
  for (int m = 16; m >= 1; m >>= 1) v += __shfl_xor(v, m, 32);
  if ((threadIdx.x & 31) == 0) atomicAdd(ntok, v);
}

// ---------------------------------------------------------------------------
// Kernel B: per centroid (h,s):
//   c = c_sum / max(count, eps)  -> bf16 table (for WMMA B operand)
//   csq[h,s] = |c|^2             -> f32 table
//   codebook += (1-g) * (|c_sum_row|^2 + count^2)   (constant term of loss)
// ---------------------------------------------------------------------------
__global__ void k_centroids(const float* __restrict__ c_sum,
                            const float* __restrict__ c_count,
                            unsigned short* __restrict__ cbf,
                            float* __restrict__ csq,
                            float* __restrict__ codebook_acc) {
  const int hs = blockIdx.x;      // 0 .. H*S-1
  const int d  = threadIdx.x;     // 0 .. 127
  float cnt = c_count[hs];
  float inv = 1.0f / fmaxf(cnt, kEps);
  float cs  = c_sum[(size_t)hs * Dn + d];
  float c   = cs * inv;
  cbf[(size_t)hs * Dn + d] = f2bf_rne(c);

  float a = c * c;      // -> csq
  float b = cs * cs;    // -> codebook constant
  #pragma unroll
  for (int m = 16; m >= 1; m >>= 1) {
    a += __shfl_xor(a, m, 32);
    b += __shfl_xor(b, m, 32);
  }
  __shared__ float sA[4], sB[4];
  int w = threadIdx.x >> 5, ln = threadIdx.x & 31;
  if (ln == 0) { sA[w] = a; sB[w] = b; }
  __syncthreads();
  if (threadIdx.x == 0) {
    float ta = sA[0] + sA[1] + sA[2] + sA[3];
    float tb = sB[0] + sB[1] + sB[2] + sB[3];
    csq[hs] = ta;
    atomicAdd(codebook_acc, kOmG * (tb + cnt * cnt));
  }
}

// ---------------------------------------------------------------------------
// Kernel C: nearest-centroid search via bf16 WMMA.
// grid = (L/128, B*H); block = 256 (8 waves). Wave w owns 16 L rows.
// Centroids staged global->LDS with async-to-LDS (ASYNCcnt), double-buffered
// 32 KB chunks of 128 centroids so chunk ck+1 streams in under chunk ck's
// 32-WMMA compute.
//
// A-fragment layout (16-bit A 16x32, ISA 7.12.2): lanes 0-15 hold row M=lane,
// K = kc*32 + {0..7, 16..23}; lanes 16-31 same row, K = kc*32 + {8..15,24..31}.
// B mirrors with N = lane&15.  C: lanes 0-15 VGPR j -> (M=j, N=lane);
// lanes 16-31 -> (M=8+j).
// ---------------------------------------------------------------------------
__global__ __launch_bounds__(256) void k_argmin(
    const float* __restrict__ vecs,
    const unsigned short* __restrict__ cbf,
    const float* __restrict__ csq,
    int* __restrict__ zout) {
  __shared__ unsigned short cL[2][128 * Dn];  // 2 x 32 KB centroid chunks
  __shared__ float vsqL[8][16];

  const int tid   = threadIdx.x;
  const int wave  = tid >> 5;
  const int lane  = tid & 31;
  const int lhalf = lane >> 4;          // 0: K-lo half, 1: K-hi half
  const int lm    = lane & 15;
  const int bh    = blockIdx.y;         // b*H + h
  const int h     = bh & (Hn - 1);
  const int l0    = blockIdx.x * 128 + wave * 16;

  const uint4* csrc = (const uint4*)(cbf + (size_t)h * Sn * Dn);
  const float* csqh = csq + (size_t)h * Sn;

  // Stage one 32KB chunk (128 centroids) into LDS buffer `buf` asynchronously.
  auto stage = [&](int ck, int buf) {
    const uint4* sp = csrc + (size_t)ck * (128 * Dn / 8);
    const unsigned ldsbase = (unsigned)(uintptr_t)(&cL[buf][0]);
    #pragma unroll
    for (int i = 0; i < 8; ++i) {
      const int e = tid + i * 256;                 // uint4 index
      async_g2l_b128(ldsbase + (unsigned)e * 16u, sp + e);
    }
  };

  // Kick chunk 0 before doing the A-fragment setup so it overlaps.
  stage(0, 0);

  // ---- Load A fragments (f32 -> bf16) + per-lane sum of squares ----
  const float* vrow = vecs + ((size_t)bh * Ln + l0 + lm) * Dn;
  union AU { unsigned short s[16]; v16bf v; };
  AU a[4];
  float sq = 0.0f;
  #pragma unroll
  for (int kc = 0; kc < 4; ++kc) {
    const int db = kc * 32 + lhalf * 8;
    const float4* p = (const float4*)(vrow + db);
    float4 f0 = p[0], f1 = p[1], f2 = p[4], f3 = p[5]; // d: db..+7, db+16..+23
    float t[16] = { f0.x, f0.y, f0.z, f0.w, f1.x, f1.y, f1.z, f1.w,
                    f2.x, f2.y, f2.z, f2.w, f3.x, f3.y, f3.z, f3.w };
    #pragma unroll
    for (int i = 0; i < 16; ++i) {
      sq += t[i] * t[i];
      a[kc].s[i] = f2bf_rne(t[i]);
    }
  }
  sq += __shfl_xor(sq, 16, 32);     // pair-lanes cover disjoint K halves
  vsqL[wave][lm] = sq;              // row (l0+lm) |v|^2  (wave-local)

  float vsq[8];
  #pragma unroll
  for (int j = 0; j < 8; ++j) vsq[j] = vsqL[wave][lhalf * 8 + j];

  float mval[8];
  int   midx[8];
  #pragma unroll
  for (int j = 0; j < 8; ++j) { mval[j] = 3.4e38f; midx[j] = 0; }

  for (int ck = 0; ck < Sn / 128; ++ck) {
    wait_async0();        // this wave's loads for chunk ck complete
    __syncthreads();      // -> everyone's chunk-ck data visible; prev compute done

    // Stream next chunk into the other buffer while we compute this one.
    if (ck + 1 < Sn / 128) stage(ck + 1, (ck + 1) & 1);

    // Prefetch the 8 |c|^2 scalars this lane needs for the chunk.
    float cq[8];
    #pragma unroll
    for (int t16 = 0; t16 < 8; ++t16)
      cq[t16] = csqh[ck * 128 + t16 * 16 + lm];

    const unsigned short* bufp = cL[ck & 1];
    #pragma unroll
    for (int t16 = 0; t16 < 8; ++t16) {
      const uint4* rowp = (const uint4*)(bufp + (t16 * 16 + lm) * Dn);
      v8f acc = {0.f, 0.f, 0.f, 0.f, 0.f, 0.f, 0.f, 0.f};
      #pragma unroll
      for (int kc = 0; kc < 4; ++kc) {
        union { uint4 q[2]; v16bf v; } bu;
        const int qb = kc * 4 + lhalf;     // uint4 = 8 bf16
        bu.q[0] = rowp[qb];                // K: kc*32 + half*8 .. +7
        bu.q[1] = rowp[qb + 2];            // K: kc*32 + half*8 + 16 ..
        acc = __builtin_amdgcn_wmma_f32_16x16x32_bf16(
            false, a[kc].v, false, bu.v, (short)0, acc, false, false);
      }
      const int sg = ck * 128 + t16 * 16 + lm;   // global centroid index
      #pragma unroll
      for (int j = 0; j < 8; ++j) {
        float d2 = vsq[j] - 2.0f * acc[j] + cq[t16];
        if (d2 < mval[j]) { mval[j] = d2; midx[j] = sg; }
      }
    }
  }

  // ---- argmin across the 16 lanes holding the same rows ----
  #pragma unroll
  for (int j = 0; j < 8; ++j) {
    #pragma unroll
    for (int m = 1; m < 16; m <<= 1) {
      float ov = __shfl_xor(mval[j], m, 32);
      int   oi = __shfl_xor(midx[j], m, 32);
      if (ov < mval[j] || (ov == mval[j] && oi < midx[j])) {
        mval[j] = ov; midx[j] = oi;
      }
    }
  }
  if (lm == 0) {
    const int rbase = l0 + lhalf * 8;
    #pragma unroll
    for (int j = 0; j < 8; ++j)
      zout[(size_t)bh * Ln + rbase + j] = midx[j];
  }
}

// ---------------------------------------------------------------------------
// Kernel D: per-token epilogue (one wave per token; 8 waves per block).
//   cz = c_sum[h,z] / max(count,eps);  vecs_hat = cz;  z as float
//   commit   += mask * |v - cz|^2 / (H * n_tok)          (exact f32)
//   codebook -= (1-g)*ND*NB * mask * (v . c_sum[h,z] + count)
// ---------------------------------------------------------------------------
__global__ __launch_bounds__(256) void k_epilogue(
    const float* __restrict__ vecs,
    const float* __restrict__ c_sum,
    const float* __restrict__ c_count,
    const int* __restrict__ mask,
    const int* __restrict__ z,
    const float* __restrict__ ntok,
    float* __restrict__ vhat,
    float* __restrict__ zf,
    float* __restrict__ commit,
    float* __restrict__ codebook) {
  const int wave = threadIdx.x >> 5;
  const int lane = threadIdx.x & 31;
  const int tok  = blockIdx.x * 8 + wave;   // 0 .. B*H*L-1
  const int bh   = tok >> 12;               // / L
  const int l    = tok & (Ln - 1);
  const int b    = bh >> 3;                 // / H
  const int h    = bh & (Hn - 1);

  const int zi = z[tok];
  const float* csr = c_sum + ((size_t)h * Sn + zi) * Dn;
  const float cnt = c_count[h * Sn + zi];
  const float inv = 1.0f / fmaxf(cnt, kEps);
  const float* vr = vecs + (size_t)tok * Dn;

  float4 cv = ((const float4*)csr)[lane];   // 4 d-elements per lane
  float4 vv = ((const float4*)vr)[lane];
  float4 cz = make_float4(cv.x * inv, cv.y * inv, cv.z * inv, cv.w * inv);
  ((float4*)(vhat + (size_t)tok * Dn))[lane] = cz;

  float e0 = vv.x - cz.x, e1 = vv.y - cz.y, e2 = vv.z - cz.z, e3 = vv.w - cz.w;
  float err2  = e0 * e0 + e1 * e1 + e2 * e2 + e3 * e3;
  float dotcs = vv.x * cv.x + vv.y * cv.y + vv.z * cv.z + vv.w * cv.w;
  #pragma unroll
  for (int m = 16; m >= 1; m >>= 1) {
    err2  += __shfl_xor(err2, m, 32);
    dotcs += __shfl_xor(dotcs, m, 32);
  }

  __shared__ float sE[8], sC[8];
  if (lane == 0) {
    zf[tok] = (float)zi;
    const float mk = (float)mask[b * Ln + l];
    sE[wave] = mk * err2;
    sC[wave] = mk * (dotcs + cnt);
  }
  __syncthreads();
  if (threadIdx.x == 0) {
    float te = 0.f, tc = 0.f;
    #pragma unroll
    for (int i = 0; i < 8; ++i) { te += sE[i]; tc += sC[i]; }
    const float nt = fmaxf(ntok[0], 1.0f);
    atomicAdd(commit, te / ((float)Hn * nt));
    atomicAdd(codebook, -kOmG * kNScl * tc);
  }
}

// ---------------------------------------------------------------------------
extern "C" void kernel_launch(void* const* d_in, const int* in_sizes, int n_in,
                              void* d_out, int out_size, void* d_ws, size_t ws_size,
                              hipStream_t stream) {
  const float* vecs    = (const float*)d_in[0];   // [B,H,L,D]
  const float* c_sum   = (const float*)d_in[1];   // [H,S,D]
  const float* c_count = (const float*)d_in[2];   // [H,S]
  const int*   lmask   = (const int*)d_in[3];     // [B,L]

  // workspace layout
  char* ws = (char*)d_ws;
  unsigned short* cbf = (unsigned short*)ws;                       // H*S*D bf16 (2 MB)
  size_t off = (size_t)Hn * Sn * Dn * sizeof(unsigned short);
  float* csq = (float*)(ws + off);                                 // H*S f32
  off += (size_t)Hn * Sn * sizeof(float);
  int* zbuf = (int*)(ws + off);                                    // B*H*L i32
  off += (size_t)Bn * Hn * Ln * sizeof(int);
  float* ntok = (float*)(ws + off);                                // 1 f32

  // output layout: vecs_hat | z(as float) | l_commit | l_codebook
  float* out    = (float*)d_out;
  float* vhat   = out;
  float* zf     = out + (size_t)Bn * Hn * Ln * Dn;
  float* commit = zf + (size_t)Bn * Hn * Ln;   // commit at [0], codebook at [1]

  k_init<<<1, 1, 0, stream>>>(commit, ntok);
  k_ntok<<<(Bn * Ln) / 256, 256, 0, stream>>>(lmask, ntok);
  k_centroids<<<Hn * Sn, 128, 0, stream>>>(c_sum, c_count, cbf, csq, commit + 1);
  k_argmin<<<dim3(Ln / 128, Bn * Hn), 256, 0, stream>>>(vecs, cbf, csq, zbuf);
  k_epilogue<<<(Bn * Hn * Ln) / 8, 256, 0, stream>>>(
      vecs, c_sum, c_count, lmask, zbuf, ntok, vhat, zf, commit, commit + 1);
}
